// AblationCohAgg_17841294148319
// MI455X (gfx1250) — matrix operand order repaired and checked by reading, not compiled, not run on hardware
//
#include <hip/hip_runtime.h>
#include <hip/hip_bf16.h>
#include <math.h>

// ---------------------------------------------------------------------------
// Problem constants (match reference)
// ---------------------------------------------------------------------------
#define NN      50000
#define IN_DIMC 1024
#define H_DIMC  256
#define Z_DIMC  64
#define NEDGE   1600000
#define ETOT    (NEDGE + NN)   // edges + self loops

typedef __attribute__((ext_vector_type(16))) __bf16 v16bf;
typedef __attribute__((ext_vector_type(8)))  float  v8f;

__device__ __forceinline__ float gelu_f(float x) {
    return 0.5f * x * (1.0f + erff(x * 0.7071067811865476f));
}

__device__ __forceinline__ void atomicMaxF32(float* addr, float val) {
    if (val >= 0.0f) atomicMax((int*)addr, __float_as_int(val));
    else             atomicMin((unsigned int*)addr, __float_as_uint(val));
}

// ---------------------------------------------------------------------------
// Weight convert + transpose: W [K x N] f32 row-major -> Wt [N x K] bf16
// ---------------------------------------------------------------------------
__global__ void k_transpose_cvt(const float* __restrict__ W, __bf16* __restrict__ Wt,
                                int K, int Ncol) {
    int i = blockIdx.x * blockDim.x + threadIdx.x;
    if (i >= K * Ncol) return;
    int n = i / K, k = i - n * K;
    Wt[(size_t)n * K + k] = (__bf16)W[(size_t)k * Ncol + n];
}

// ---------------------------------------------------------------------------
// WMMA GEMM with N-register-blocking:
//   Each wave computes a 16 x (16*NT) output strip. The A fragment for a
//   K-step is loaded ONCE and swept against NT B fragments (NT accumulators).
//   => streaming input A is read exactly once from HBM when 16*NT == Ncol.
//   A: f32 (converted to bf16 in regs), Bt: bf16 pre-transposed [Ncol][K]
// ---------------------------------------------------------------------------
template<bool BIAS, bool ACT, int NT>
__global__ void k_wmma_gemm(const float* __restrict__ A, const __bf16* __restrict__ Bt,
                            const float* __restrict__ bias, float* __restrict__ C,
                            int M, int K, int Ncol) {
    int gid    = blockIdx.x * blockDim.x + threadIdx.x;
    int wave   = gid >> 5;
    int lane   = threadIdx.x & 31;
    int strips = Ncol / (16 * NT);
    int total  = (M >> 4) * strips;
    if (wave >= total) return;
    int mt = wave / strips, st = wave - mt * strips;
    int m0 = mt << 4, n0 = st * (16 * NT);
    int lm = lane & 15, lh = lane >> 4;

    v8f acc[NT];
    #pragma unroll
    for (int t = 0; t < NT; ++t) acc[t] = (v8f){};

    const float*  arow  = A  + (size_t)(m0 + lm) * K;
    const __bf16* bbase = Bt + (size_t)(n0 + lm) * K;

    for (int k0 = 0; k0 < K; k0 += 32) {
        // A fragment: lanes 0-15 hold chunk-K 0..15, lanes 16-31: 16..31
        v16bf a;
        const float* ap0 = arow + k0 + lh * 16;      // elements 0..7
        const float* ap1 = ap0 + 8;                  // elements 8..15
        __builtin_prefetch(ap0 + 32, 0, 0);          // next K-step of the A stream
        #pragma unroll
        for (int i = 0; i < 8; ++i) {
            a[i]     = (__bf16)ap0[i];
            a[8 + i] = (__bf16)ap1[i];
        }
        // Sweep NT B fragments (L2-resident weights) against this A fragment
        #pragma unroll
        for (int t = 0; t < NT; ++t) {
            v16bf b = *(const v16bf*)(bbase + (size_t)t * 16 * K + k0 + lh * 16);
            acc[t] = __builtin_amdgcn_wmma_f32_16x16x32_bf16(false, a, false, b,
                                                             (short)0, acc[t], false, false);
        }
    }

    #pragma unroll
    for (int t = 0; t < NT; ++t) {
        int col = n0 + t * 16 + lm;
        float bv = BIAS ? bias[col] : 0.0f;
        #pragma unroll
        for (int r = 0; r < 8; ++r) {
            int row = m0 + lh * 8 + r;   // C layout: VGPR r -> M = r (+8 for hi lanes)
            float v = acc[t][r] + bv;
            if (ACT) v = gelu_f(v);
            C[(size_t)row * Ncol + col] = v;
        }
    }
}

// ---------------------------------------------------------------------------
// Fused decoder GEMM + MSE accumulation:
//   Xhat = Z[M,64] @ Wd[64,1024] + bd ; acc += sum((Xhat - X)^2)
// Never materializes Xhat (saves ~410 MB of HBM traffic). Z (12.8 MB) and
// Wd (128 KB bf16) are L2-resident, so per-tile re-reads are cheap.
// ---------------------------------------------------------------------------
__global__ void k_decoder_loss(const float* __restrict__ Z, const __bf16* __restrict__ Bt,
                               const float* __restrict__ bias, const float* __restrict__ X,
                               double* __restrict__ acc) {
    const int K = Z_DIMC, Ncol = IN_DIMC, M = NN;
    int gid    = blockIdx.x * blockDim.x + threadIdx.x;
    int wave   = gid >> 5;
    int lane   = threadIdx.x & 31;
    int tilesN = Ncol >> 4;
    int total  = (M >> 4) * tilesN;
    if (wave >= total) return;
    int mt = wave / tilesN, nt = wave - mt * tilesN;
    int m0 = mt << 4, n0 = nt << 4;
    int lm = lane & 15, lh = lane >> 4;

    v8f c = {};
    const float*  arow = Z  + (size_t)(m0 + lm) * K;
    const __bf16* brow = Bt + (size_t)(n0 + lm) * K;
    #pragma unroll
    for (int k0 = 0; k0 < K; k0 += 32) {
        v16bf a;
        const float* ap0 = arow + k0 + lh * 16;
        const float* ap1 = ap0 + 8;
        #pragma unroll
        for (int i = 0; i < 8; ++i) { a[i] = (__bf16)ap0[i]; a[8 + i] = (__bf16)ap1[i]; }
        v16bf b = *(const v16bf*)(brow + k0 + lh * 16);
        c = __builtin_amdgcn_wmma_f32_16x16x32_bf16(false, a, false, b,
                                                    (short)0, c, false, false);
    }

    int col = n0 + lm;
    float bv = bias[col];
    float part = 0.0f;
    #pragma unroll
    for (int r = 0; r < 8; ++r) {
        int row = m0 + lh * 8 + r;
        float d = c[r] + bv - X[(size_t)row * Ncol + col];
        part += d * d;
    }
    for (int off = 16; off > 0; off >>= 1) part += __shfl_down(part, off, 32);
    if (lane == 0) atomicAdd(acc, (double)part);
}

// ---------------------------------------------------------------------------
// Per-node attention scalars: as[i] = h[i]·a_src ; ad[i] = h[i]·a_dst
// One wave per node: coalesced row read + wave32 reduction.
// ---------------------------------------------------------------------------
__global__ void k_alpha(const float* __restrict__ H, const float* __restrict__ a_s,
                        const float* __restrict__ a_d, float* __restrict__ as,
                        float* __restrict__ ad) {
    int wave = (blockIdx.x * blockDim.x + threadIdx.x) >> 5;
    int lane = threadIdx.x & 31;
    if (wave >= NN) return;
    const float* h = H + (size_t)wave * Z_DIMC;
    float h0 = h[lane], h1 = h[lane + 32];
    float s = h0 * a_s[lane] + h1 * a_s[lane + 32];
    float d = h0 * a_d[lane] + h1 * a_d[lane + 32];
    for (int off = 16; off > 0; off >>= 1) {
        s += __shfl_down(s, off, 32);
        d += __shfl_down(d, off, 32);
    }
    if (lane == 0) { as[wave] = s; ad[wave] = d; }
}

__global__ void k_fill(float* __restrict__ p, float v, int n) {
    int i = blockIdx.x * blockDim.x + threadIdx.x;
    if (i < n) p[i] = v;
}

// Pass 1: e = leaky_relu(as[src]+ad[dst]); segment max over dst
__global__ void k_edge1(const int* __restrict__ src, const int* __restrict__ dst,
                        const float* __restrict__ as, const float* __restrict__ ad,
                        float* __restrict__ e, float* __restrict__ mx) {
    int eid = blockIdx.x * blockDim.x + threadIdx.x;
    if (eid >= ETOT) return;
    int s, d;
    if (eid < NEDGE) { s = src[eid]; d = dst[eid]; } else { s = d = eid - NEDGE; }
    float x = as[s] + ad[d];
    x = (x > 0.0f) ? x : 0.2f * x;
    e[eid] = x;
    atomicMaxF32(mx + d, x);
}

// Pass 2: p = exp(e - mx[dst]); segment sum over dst
__global__ void k_edge2(const int* __restrict__ dst, const float* __restrict__ e,
                        const float* __restrict__ mx, float* __restrict__ p,
                        float* __restrict__ sm) {
    int eid = blockIdx.x * blockDim.x + threadIdx.x;
    if (eid >= ETOT) return;
    int d = (eid < NEDGE) ? dst[eid] : (eid - NEDGE);
    float pv = __expf(e[eid] - mx[d]);
    p[eid] = pv;
    atomicAdd(sm + d, pv);
}

// Pass 3: agg[dst] += h[src] * (p / sm[dst]) over 64 channels
__global__ void k_edge3(const int* __restrict__ src, const int* __restrict__ dst,
                        const float* __restrict__ p, const float* __restrict__ sm,
                        const float* __restrict__ h, float* __restrict__ agg) {
    long long tid = (long long)blockIdx.x * blockDim.x + threadIdx.x;
    if (tid >= (long long)ETOT * Z_DIMC) return;
    int eid = (int)(tid >> 6);
    int ch  = (int)(tid & 63);
    int s, d;
    if (eid < NEDGE) { s = src[eid]; d = dst[eid]; } else { s = d = eid - NEDGE; }
    float coef = p[eid] / sm[d];
    atomicAdd(agg + (size_t)d * Z_DIMC + ch, h[(size_t)s * Z_DIMC + ch] * coef);
}

// z = gelu(agg + b), in place, [NN x 64]
__global__ void k_biasgelu(float* __restrict__ z, const float* __restrict__ b) {
    int i = blockIdx.x * blockDim.x + threadIdx.x;
    if (i >= NN * Z_DIMC) return;
    z[i] = gelu_f(z[i] + b[i & (Z_DIMC - 1)]);
}

__global__ void k_zero_acc(double* acc) { if (threadIdx.x == 0 && blockIdx.x == 0) acc[0] = 0.0; }

__global__ void k_finalize(const double* __restrict__ acc, float* __restrict__ out) {
    if (threadIdx.x == 0 && blockIdx.x == 0)
        out[0] = (float)(acc[0] / ((double)NN * (double)IN_DIMC));
}

// ---------------------------------------------------------------------------
// Host-side orchestration
// ---------------------------------------------------------------------------
static inline size_t alignup(size_t x) { return (x + 255) & ~(size_t)255; }

extern "C" void kernel_launch(void* const* d_in, const int* in_sizes, int n_in,
                              void* d_out, int out_size, void* d_ws, size_t ws_size,
                              hipStream_t stream) {
    (void)in_sizes; (void)n_in; (void)out_size; (void)ws_size;
    const float* X   = (const float*)d_in[0];
    const int*   ei  = (const int*)  d_in[1];
    const float* W1  = (const float*)d_in[2];
    const float* b1  = (const float*)d_in[3];
    const float* W2  = (const float*)d_in[4];
    const float* b2  = (const float*)d_in[5];
    const float* g1W = (const float*)d_in[6];
    const float* g1as= (const float*)d_in[7];
    const float* g1ad= (const float*)d_in[8];
    const float* g1b = (const float*)d_in[9];
    const float* g2W = (const float*)d_in[10];
    const float* g2as= (const float*)d_in[11];
    const float* g2ad= (const float*)d_in[12];
    const float* g2b = (const float*)d_in[13];
    const float* Wl  = (const float*)d_in[14];
    const float* bl  = (const float*)d_in[15];
    const float* Wg  = (const float*)d_in[16];
    const float* bg  = (const float*)d_in[17];
    const float* Wd  = (const float*)d_in[18];
    const float* bd  = (const float*)d_in[19];
    float* out = (float*)d_out;

    const int* srcp = ei;
    const int* dstp = ei + NEDGE;

    // ---- workspace layout (bytes) ----
    char* base = (char*)d_ws;
    size_t o = 0;
    __bf16* w1t = (__bf16*)(base + o); o = alignup(o + (size_t)IN_DIMC * H_DIMC * 2);
    __bf16* w2t = (__bf16*)(base + o); o = alignup(o + (size_t)H_DIMC * H_DIMC * 2);
    __bf16* g1t = (__bf16*)(base + o); o = alignup(o + (size_t)H_DIMC * Z_DIMC * 2);
    __bf16* g2t = (__bf16*)(base + o); o = alignup(o + (size_t)Z_DIMC * Z_DIMC * 2);
    __bf16* wlt = (__bf16*)(base + o); o = alignup(o + (size_t)Z_DIMC * Z_DIMC * 2);
    __bf16* wgt = (__bf16*)(base + o); o = alignup(o + (size_t)Z_DIMC * Z_DIMC * 2);
    __bf16* wdt = (__bf16*)(base + o); o = alignup(o + (size_t)Z_DIMC * IN_DIMC * 2);

    const size_t ZBYTES = (size_t)NN * Z_DIMC * 4;       // 12.8 MB
    float* H1   = (float*)(base + o); o = alignup(o + (size_t)NN * H_DIMC * 4);  // 51.2 MB
    float* H2   = (float*)(base + o); o = alignup(o + (size_t)NN * H_DIMC * 4);  // 51.2 MB
    // H1 region recycled after GEMM2 (exactly 4 x 12.8 MB):
    float* hb1  = (float*)((char*)H1 + 0 * ZBYTES);
    float* agg1 = (float*)((char*)H1 + 1 * ZBYTES);
    float* hb2  = (float*)((char*)H1 + 2 * ZBYTES);
    float* agg2 = (float*)((char*)H1 + 3 * ZBYTES);
    // H2 region recycled after GEMM3:
    float* zl   = (float*)((char*)H2 + 0 * ZBYTES);
    float* zg   = (float*)((char*)H2 + 1 * ZBYTES);

    float* asb = (float*)(base + o); o = alignup(o + (size_t)NN * 4);
    float* adb = (float*)(base + o); o = alignup(o + (size_t)NN * 4);
    float* mxb = (float*)(base + o); o = alignup(o + (size_t)NN * 4);
    float* smb = (float*)(base + o); o = alignup(o + (size_t)NN * 4);
    float* eb  = (float*)(base + o); o = alignup(o + (size_t)ETOT * 4);
    float* pb  = (float*)(base + o); o = alignup(o + (size_t)ETOT * 4);
    double* acc = (double*)(base + o); o = alignup(o + 8);

    const int TB = 256;
    auto cdiv = [](long long a, long long b) { return (int)((a + b - 1) / b); };

    // 1) weight convert+transpose (tiny, once per launch)
    k_transpose_cvt<<<cdiv(IN_DIMC * H_DIMC, TB), TB, 0, stream>>>(W1, w1t, IN_DIMC, H_DIMC);
    k_transpose_cvt<<<cdiv(H_DIMC * H_DIMC, TB), TB, 0, stream>>>(W2, w2t, H_DIMC, H_DIMC);
    k_transpose_cvt<<<cdiv(H_DIMC * Z_DIMC, TB), TB, 0, stream>>>(g1W, g1t, H_DIMC, Z_DIMC);
    k_transpose_cvt<<<cdiv(Z_DIMC * Z_DIMC, TB), TB, 0, stream>>>(g2W, g2t, Z_DIMC, Z_DIMC);
    k_transpose_cvt<<<cdiv(Z_DIMC * Z_DIMC, TB), TB, 0, stream>>>(Wl, wlt, Z_DIMC, Z_DIMC);
    k_transpose_cvt<<<cdiv(Z_DIMC * Z_DIMC, TB), TB, 0, stream>>>(Wg, wgt, Z_DIMC, Z_DIMC);
    k_transpose_cvt<<<cdiv(Z_DIMC * IN_DIMC, TB), TB, 0, stream>>>(Wd, wdt, Z_DIMC, IN_DIMC);

    // 2) encoder MLP: one wave per 16x256 row strip => X and H1 read once
    {
        int waves = NN / 16;   // strips = 256/(16*16) = 1
        k_wmma_gemm<true, true, 16><<<cdiv(waves, 8), TB, 0, stream>>>(X, w1t, b1, H1, NN, IN_DIMC, H_DIMC);
        k_wmma_gemm<true, true, 16><<<cdiv(waves, 8), TB, 0, stream>>>(H1, w2t, b2, H2, NN, H_DIMC, H_DIMC);
    }

    // 3) GAT layers
    auto gat = [&](const float* zin, const __bf16* wt, int Kdim,
                   const float* a_s, const float* a_d, const float* bvec,
                   float* hbuf, float* aggbuf) {
        int wv = NN / 16;      // strips = 64/(16*4) = 1
        k_wmma_gemm<false, false, 4><<<cdiv(wv, 8), TB, 0, stream>>>(zin, wt, nullptr, hbuf, NN, Kdim, Z_DIMC);
        k_alpha<<<cdiv((long long)NN * 32, TB), TB, 0, stream>>>(hbuf, a_s, a_d, asb, adb);
        k_fill<<<cdiv(NN, TB), TB, 0, stream>>>(mxb, -INFINITY, NN);
        k_fill<<<cdiv(NN, TB), TB, 0, stream>>>(smb, 0.0f, NN);
        k_fill<<<cdiv((long long)NN * Z_DIMC, TB), TB, 0, stream>>>(aggbuf, 0.0f, NN * Z_DIMC);
        k_edge1<<<cdiv(ETOT, TB), TB, 0, stream>>>(srcp, dstp, asb, adb, eb, mxb);
        k_edge2<<<cdiv(ETOT, TB), TB, 0, stream>>>(dstp, eb, mxb, pb, smb);
        k_edge3<<<cdiv((long long)ETOT * Z_DIMC, TB), TB, 0, stream>>>(srcp, dstp, pb, smb, hbuf, aggbuf);
        k_biasgelu<<<cdiv((long long)NN * Z_DIMC, TB), TB, 0, stream>>>(aggbuf, bvec);
    };

    gat(H2,   g1t, H_DIMC, g1as, g1ad, g1b, hb1, agg1);   // z1 = agg1
    gat(agg1, g2t, Z_DIMC, g2as, g2ad, g2b, hb2, agg2);   // z2 = agg2

    // 4) small linears: zl = z2@Wl+bl ; zg = zl@Wg+bg
    {
        int wv = NN / 16;
        k_wmma_gemm<true, false, 4><<<cdiv(wv, 8), TB, 0, stream>>>(agg2, wlt, bl, zl, NN, Z_DIMC, Z_DIMC);
        k_wmma_gemm<true, false, 4><<<cdiv(wv, 8), TB, 0, stream>>>(zl,   wgt, bg, zg, NN, Z_DIMC, Z_DIMC);
    }

    // 5) fused decoder + MSE
    k_zero_acc<<<1, 1, 0, stream>>>(acc);
    {
        int wv = (NN / 16) * (IN_DIMC / 16);
        k_decoder_loss<<<cdiv(wv, 8), TB, 0, stream>>>(zg, wdt, bd, X, acc);
    }
    k_finalize<<<1, 1, 0, stream>>>(acc, out);
}
